// BaselineASGCN_36120674959481
// MI455X (gfx1250) — compile-verified, matrix operand order checked by reading, not run on hardware
//
#include <hip/hip_runtime.h>
#include <hip/hip_bf16.h>

typedef __attribute__((ext_vector_type(16))) __bf16 v16bf;
typedef __attribute__((ext_vector_type(8)))  float  v8f;

#define N_NODES_C   100000
#define N_EDGES_C   1600000
#define IN_DIM_C    300
#define KPAD1_C     320
#define HID_C       128
#define N_ASPECTS_C 2048
#define N_CLASSES_C 3

union FragBF { uint4 u[2]; v16bf v; };

// ---------------- degree / norm ----------------

__global__ void init_deg_kernel(float* deg, int n) {
    int i = blockIdx.x * blockDim.x + threadIdx.x;
    if (i < n) deg[i] = 1.0f;   // self loop contributes 1
}

__global__ void edge_deg_kernel(const long long* __restrict__ dst, float* deg, int nEdges) {
    int e = blockIdx.x * blockDim.x + threadIdx.x;
    if (e < nEdges) atomicAdd(&deg[(int)dst[e]], 1.0f);
}

__global__ void rsqrt_kernel(float* deg, int n) {
    int i = blockIdx.x * blockDim.x + threadIdx.x;
    if (i < n) deg[i] = rsqrtf(deg[i]);   // deg >= 1 always (self loop)
}

// ---------------- bf16 staging ----------------

// row-major [M x Kin] f32 -> row-major [M x Kpad] bf16, zero-padded K
__global__ void pad_cast_bf16_kernel(const float* __restrict__ in, __bf16* __restrict__ out,
                                     int Kin, int Kpad, long long total) {
    long long idx = (long long)blockIdx.x * blockDim.x + threadIdx.x;
    if (idx >= total) return;
    int k = (int)(idx % Kpad);
    long long r = idx / Kpad;
    out[idx] = (k < Kin) ? (__bf16)in[r * Kin + k] : (__bf16)0.0f;
}

// W [Kin x Ncols] f32 row-major -> Wt [Ncols x Kpad] bf16 (transposed, padded)
__global__ void transpose_pad_bf16_kernel(const float* __restrict__ W, __bf16* __restrict__ Wt,
                                          int Kin, int Kpad, int Ncols) {
    int idx = blockIdx.x * blockDim.x + threadIdx.x;
    int total = Ncols * Kpad;
    if (idx >= total) return;
    int n = idx / Kpad, k = idx % Kpad;
    Wt[(long long)n * Kpad + k] = (k < Kin) ? (__bf16)W[(long long)k * Ncols + n] : (__bf16)0.0f;
}

// ---------------- WMMA GEMM:  C[M x 128] = A[M x K] * Bt[128 x K]^T ----------------

template<int KTOT>
__global__ void wmma_gemm_nt_kernel(const __bf16* __restrict__ A,
                                    const __bf16* __restrict__ Bt,
                                    float* __restrict__ C, int M) {
    const int NCOL = 128;
    int wave = (int)((blockIdx.x * blockDim.x + threadIdx.x) >> 5);
    int lane = threadIdx.x & 31;
    int row0 = wave * 16;
    if (row0 >= M) return;                 // wave-uniform: EXEC all-ones at WMMA

    int laneHi = lane >> 4;                // 0|1 -> K halves per ISA A/B fragment layout
    int laneLo = lane & 15;
    long long arow = (long long)(row0 + laneLo) * KTOT;

    v8f zero = {0.f, 0.f, 0.f, 0.f, 0.f, 0.f, 0.f, 0.f};
    v8f acc[8];
#pragma unroll
    for (int t = 0; t < 8; ++t) acc[t] = zero;

    for (int k = 0; k < KTOT; k += 32) {
        int kb = k + laneHi * 8;
        FragBF a;
        a.u[0] = *(const uint4*)(A + arow + kb);        // K = kb .. kb+7
        a.u[1] = *(const uint4*)(A + arow + kb + 16);   // K = kb+16 .. kb+23
#pragma unroll
        for (int t = 0; t < 8; ++t) {
            long long brow = (long long)(t * 16 + laneLo) * KTOT;
            FragBF b;
            b.u[0] = *(const uint4*)(Bt + brow + kb);
            b.u[1] = *(const uint4*)(Bt + brow + kb + 16);
            acc[t] = __builtin_amdgcn_wmma_f32_16x16x32_bf16(
                false, a.v, false, b.v, (short)0, acc[t], false, false);
        }
    }

    // C/D layout: VGPR j -> row (row0 + j + 8*laneHi), col = t*16 + laneLo
#pragma unroll
    for (int t = 0; t < 8; ++t) {
        int col = t * 16 + laneLo;
#pragma unroll
        for (int j = 0; j < 8; ++j) {
            C[(long long)(row0 + j + 8 * laneHi) * NCOL + col] = acc[t][j];
        }
    }
}

// ---------------- aggregation (self loop + bias, then edge scatter) ----------------

__global__ void self_bias_kernel(const float* __restrict__ h, const float* __restrict__ dinv,
                                 const float* __restrict__ bias, float* __restrict__ agg,
                                 long long total) {
    long long idx = (long long)blockIdx.x * blockDim.x + threadIdx.x;
    if (idx >= total) return;
    int c = (int)(idx & (HID_C - 1));
    long long i = idx >> 7;   // / 128
    float d = dinv[i];
    agg[idx] = h[idx] * d * d + bias[c];
}

// one wave per edge; each lane handles 4 channels (32*4 = 128)
__global__ void edge_scatter_kernel(const long long* __restrict__ src,
                                    const long long* __restrict__ dst,
                                    const float* __restrict__ h,
                                    const float* __restrict__ dinv,
                                    float* __restrict__ agg, int nEdges) {
    int wave = (int)((blockIdx.x * blockDim.x + threadIdx.x) >> 5);
    int lane = threadIdx.x & 31;
    if (wave >= nEdges) return;
    int s = (int)src[wave];
    int d = (int)dst[wave];
    float w = dinv[s] * dinv[d];
    const float4* hp = (const float4*)(h + (long long)s * HID_C);
    float4 v = hp[lane];
    float* ap = agg + (long long)d * HID_C + lane * 4;
    atomicAdd(ap + 0, v.x * w);
    atomicAdd(ap + 1, v.y * w);
    atomicAdd(ap + 2, v.z * w);
    atomicAdd(ap + 3, v.w * w);
}

__global__ void relu_cast_bf16_kernel(const float* __restrict__ in, __bf16* __restrict__ out,
                                      long long total) {
    long long idx = (long long)blockIdx.x * blockDim.x + threadIdx.x;
    if (idx >= total) return;
    out[idx] = (__bf16)fmaxf(in[idx], 0.0f);
}

// ---------------- classifier ----------------

__global__ void classify_kernel(const float* __restrict__ h, const long long* __restrict__ aidx,
                                const float* __restrict__ Wc, const float* __restrict__ bc,
                                float* __restrict__ out, int nAsp) {
    int idx = blockIdx.x * blockDim.x + threadIdx.x;
    if (idx >= nAsp * N_CLASSES_C) return;
    int a = idx / N_CLASSES_C;
    int j = idx % N_CLASSES_C;
    const float* row = h + (long long)aidx[a] * HID_C;
    float s = bc[j];
#pragma unroll 4
    for (int c = 0; c < HID_C; ++c) s += row[c] * Wc[c * N_CLASSES_C + j];
    out[idx] = s;
}

// ---------------- launch ----------------

extern "C" void kernel_launch(void* const* d_in, const int* in_sizes, int n_in,
                              void* d_out, int out_size, void* d_ws, size_t ws_size,
                              hipStream_t stream) {
    const float*     features = (const float*)d_in[0];
    const long long* edge_idx = (const long long*)d_in[1];
    const long long* asp_idx  = (const long long*)d_in[2];
    const float*     W1       = (const float*)d_in[3];
    const float*     b1       = (const float*)d_in[4];
    const float*     W2       = (const float*)d_in[5];
    const float*     b2       = (const float*)d_in[6];
    const float*     Wc       = (const float*)d_in[7];
    const float*     bc       = (const float*)d_in[8];
    float*           out      = (float*)d_out;

    const long long* e_src = edge_idx;
    const long long* e_dst = edge_idx + N_EDGES_C;

    // workspace carve-up (256B aligned), with reuse
    char* ws = (char*)d_ws;
    size_t off = 0;
    auto carve = [&](size_t bytes) { char* p = ws + off; off = (off + bytes + 255) & ~(size_t)255; return p; };

    float*  dinv = (float*)carve((size_t)N_NODES_C * 4);                    // deg -> rsqrt in place
    __bf16* Xb   = (__bf16*)carve((size_t)N_NODES_C * KPAD1_C * 2);         // reused as X2b later
    __bf16* W1t  = (__bf16*)carve((size_t)HID_C * KPAD1_C * 2);
    __bf16* W2t  = (__bf16*)carve((size_t)HID_C * HID_C * 2);
    float*  h    = (float*)carve((size_t)N_NODES_C * HID_C * 4);            // h1 then h2
    float*  agg  = (float*)carve((size_t)N_NODES_C * HID_C * 4);            // agg1 then agg2
    __bf16* X2b  = Xb;                                                       // reuse (Xb dead after GEMM1)

    const int T = 256;
    long long nXpad = (long long)N_NODES_C * KPAD1_C;
    long long nH    = (long long)N_NODES_C * HID_C;

    // --- normalization ---
    init_deg_kernel<<<(N_NODES_C + T - 1) / T, T, 0, stream>>>(dinv, N_NODES_C);
    edge_deg_kernel<<<(N_EDGES_C + T - 1) / T, T, 0, stream>>>(e_dst, dinv, N_EDGES_C);
    rsqrt_kernel<<<(N_NODES_C + T - 1) / T, T, 0, stream>>>(dinv, N_NODES_C);

    // --- stage bf16 operands ---
    pad_cast_bf16_kernel<<<(int)((nXpad + T - 1) / T), T, 0, stream>>>(features, Xb, IN_DIM_C, KPAD1_C, nXpad);
    transpose_pad_bf16_kernel<<<(HID_C * KPAD1_C + T - 1) / T, T, 0, stream>>>(W1, W1t, IN_DIM_C, KPAD1_C, HID_C);
    transpose_pad_bf16_kernel<<<(HID_C * HID_C + T - 1) / T, T, 0, stream>>>(W2, W2t, HID_C, HID_C, HID_C);

    // --- layer 1: WMMA GEMM + norm-aggregate + relu ---
    int gemmThreads = (N_NODES_C / 16) * 32;   // one wave per 16-row stripe
    wmma_gemm_nt_kernel<KPAD1_C><<<(gemmThreads + T - 1) / T, T, 0, stream>>>(Xb, W1t, h, N_NODES_C);
    self_bias_kernel<<<(int)((nH + T - 1) / T), T, 0, stream>>>(h, dinv, b1, agg, nH);
    edge_scatter_kernel<<<(N_EDGES_C * 32 + T - 1) / T, T, 0, stream>>>(e_src, e_dst, h, dinv, agg, N_EDGES_C);
    relu_cast_bf16_kernel<<<(int)((nH + T - 1) / T), T, 0, stream>>>(agg, X2b, nH);

    // --- layer 2: WMMA GEMM + norm-aggregate ---
    wmma_gemm_nt_kernel<HID_C><<<(gemmThreads + T - 1) / T, T, 0, stream>>>(X2b, W2t, h, N_NODES_C);
    self_bias_kernel<<<(int)((nH + T - 1) / T), T, 0, stream>>>(h, dinv, b2, agg, nH);
    edge_scatter_kernel<<<(N_EDGES_C * 32 + T - 1) / T, T, 0, stream>>>(e_src, e_dst, h, dinv, agg, N_EDGES_C);

    // --- classifier head ---
    classify_kernel<<<(N_ASPECTS_C * N_CLASSES_C + T - 1) / T, T, 0, stream>>>(agg, asp_idx, Wc, bc, out, N_ASPECTS_C);
}
